// NestedFC_2448131359320
// MI455X (gfx1250) — matrix operand order repaired and checked by reading, not compiled, not run on hardware
//
#include <hip/hip_runtime.h>

#define B_   2048
#define E_   32
#define IN_  2048
#define OUT_ 1024
#define TOPK 4

typedef __attribute__((ext_vector_type(16))) __bf16 v16bf;
typedef __attribute__((ext_vector_type(8)))  __bf16 v8bf;
typedef __attribute__((ext_vector_type(4)))  __bf16 v4bf;
typedef __attribute__((ext_vector_type(8)))  float  v8f;
typedef __attribute__((ext_vector_type(4)))  float  v4f;

union Frag16 {
    v16bf v;
    v8bf  h[2];
    uint4 q[2];
};

// Workgroup split barrier, producer side: make this wave's LDS stores visible
// (drain DScnt only -- outstanding *global loads* may stay in flight, which is
// what lets the register-pipelined prefetch cross the barrier).
__device__ __forceinline__ void wg_barrier_stores_visible() {
    asm volatile("s_wait_dscnt 0x0\n\t"
                 "s_barrier_signal -1\n\t"
                 "s_barrier_wait -1" ::: "memory");
}

// Workgroup split barrier, consumer side: drain DScnt before signaling so all
// of this wave's fragment ds_loads have retired (data safely in VGPRs) before
// any other wave may overwrite the LDS tile. Global loads stay in flight.
__device__ __forceinline__ void wg_barrier_reads_done() {
    asm volatile("s_wait_dscnt 0x0\n\t"
                 "s_barrier_signal -1\n\t"
                 "s_barrier_wait -1" ::: "memory");
}

__global__ void zero_counts(int* counts) {
    if (threadIdx.x < E_) counts[threadIdx.x] = 0;
}

// One thread per sample: pick 4 smallest activations (stable ascending),
// append packed (sample<<2 | slot) to each chosen expert's bucket.
__global__ void route_kernel(const float* __restrict__ act,
                             int* __restrict__ counts,
                             int* __restrict__ pairs) {
    int b = blockIdx.x * blockDim.x + threadIdx.x;
    if (b >= B_) return;
    float v[E_];
#pragma unroll
    for (int e = 0; e < E_; ++e) v[e] = act[b * E_ + e];
    unsigned used = 0u;
#pragma unroll
    for (int k = 0; k < TOPK; ++k) {
        int best = -1;
        float bestv = 0.0f;
#pragma unroll
        for (int e = 0; e < E_; ++e) {
            if (used & (1u << e)) continue;
            if (best < 0 || v[e] < bestv) { best = e; bestv = v[e]; }
        }
        used |= 1u << best;
        int pos = atomicAdd(&counts[best], 1);
        pairs[best * B_ + pos] = (b << 2) | k;
    }
}

__device__ __forceinline__ void store_tile(const v8f& a, int mrow0, int hh, int mlen,
                                           const int* sPairs, float* __restrict__ out,
                                           int outcol, float bv) {
#pragma unroll
    for (int vv = 0; vv < 8; ++vv) {
        int row = mrow0 + hh * 8 + vv;
        if (row < mlen) {
            int p = sPairs[row];
            int sample = p >> 2;
            int slot   = p & 3;
            out[((size_t)sample * TOPK + slot) * OUT_ + outcol] = a[vv] + bv;
        }
    }
}

// Block = 512 threads (16 waves). Block tile: 256 gathered samples x 64 OUT
// cols for one expert. Per K-step: A (256x32) and B (64x32) converted f32->bf16
// and staged in LDS; wave (nsub, mq) owns cols nsub*16..+15 and rows
// mq*64..+63 -> 4x v_wmma_f32_16x16x32_bf16. Next K-step's global loads are
// pipelined through registers and stay in flight across the raw split
// barriers (only DScnt is drained), hiding HBM/L2 latency behind the WMMAs.
__global__ __launch_bounds__(512)
void moe_gemm(const float* __restrict__ feat, const float* __restrict__ W,
              const float* __restrict__ bias, const int* __restrict__ counts,
              const int* __restrict__ pairs, float* __restrict__ out) {
    const int e     = blockIdx.x;
    const int n0    = blockIdx.y * 64;
    const int mbase = blockIdx.z * 256;
    const int cnt   = counts[e];
    if (mbase >= cnt) return;
    const int mlen = min(256, cnt - mbase);

    __shared__ unsigned short sA[256 * 40];  // 256 rows x 32 bf16, stride 40
    __shared__ unsigned short sB[64 * 40];   // 64 cols  x 32 bf16, stride 40
    __shared__ int sPairs[256];

    const int t = threadIdx.x;
    if (t < 256) {
        sPairs[t] = (mbase + t < cnt) ? pairs[e * B_ + mbase + t] : 0;
    }
    __syncthreads();

    const int lane = t & 31;
    const int wv   = t >> 5;        // 0..15
    const int nsub = wv & 3;        // N subtile (16 cols)
    const int mq   = wv >> 2;       // M quarter (64 rows)
    const int r    = lane & 15;
    const int hh   = lane >> 4;

    // A staging: each thread converts half a row (16 floats)
    const int srow = t >> 1;            // 0..255
    const int soff = (t & 1) * 16;      // 0 or 16
    const float* arow = feat + (size_t)(sPairs[srow] >> 2) * IN_;

    // B staging: each thread converts 4 floats of one W row
    const int bcol = t >> 3;            // 0..63
    const int bseg = (t & 7) * 4;       // 0..28
    const float* bsrc = W + ((size_t)e * OUT_ + (size_t)(n0 + bcol)) * (size_t)IN_;

    // Fragment read bases (halfword offsets)
    const unsigned short* aF = &sA[(mq * 64 + r) * 40 + hh * 8];
    const unsigned short* bF = &sB[(nsub * 16 + r) * 40 + hh * 8];

    v8f acc0 = {}, acc1 = {}, acc2 = {}, acc3 = {};

    // Prologue: prefetch K-step 0 into registers.
    v8f ar0 = *(const v8f*)(arow + soff);
    v8f ar1 = *(const v8f*)(arow + soff + 8);
    v4f br  = *(const v4f*)(bsrc + bseg);

    for (int k0 = 0; k0 < IN_; k0 += 32) {
        {   // convert current prefetched registers -> LDS
            Frag16 fr;
            fr.h[0] = __builtin_convertvector(ar0, v8bf);
            fr.h[1] = __builtin_convertvector(ar1, v8bf);
            uint4* dst = (uint4*)&sA[srow * 40 + soff];
            dst[0] = fr.q[0];
            dst[1] = fr.q[1];
            *(v4bf*)&sB[bcol * 40 + bseg] = __builtin_convertvector(br, v4bf);
        }

        wg_barrier_stores_visible();   // DScnt drain only

        // Issue next K-step's global loads; they remain in flight across the
        // barriers and are consumed at the top of the next iteration.
        const int kn = (k0 + 32 < IN_) ? (k0 + 32) : k0;
        ar0 = *(const v8f*)(arow + kn + soff);
        ar1 = *(const v8f*)(arow + kn + soff + 8);
        br  = *(const v4f*)(bsrc + kn + bseg);

        Frag16 fb, fa0, fa1, fa2, fa3;
        // B (32x16): elems 0-7 = K hh*8+j, elems 8-15 = K 16+hh*8+j
        fb.q[0] = *(const uint4*)(bF);
        fb.q[1] = *(const uint4*)(bF + 16);
        // A (16x32): elems 0-7 = K hh*8+j, elems 8-15 = K 16+hh*8+j
        fa0.q[0] = *(const uint4*)(aF);
        fa0.q[1] = *(const uint4*)(aF + 16);
        fa1.q[0] = *(const uint4*)(aF + 640);
        fa1.q[1] = *(const uint4*)(aF + 656);
        fa2.q[0] = *(const uint4*)(aF + 1280);
        fa2.q[1] = *(const uint4*)(aF + 1296);
        fa3.q[0] = *(const uint4*)(aF + 1920);
        fa3.q[1] = *(const uint4*)(aF + 1936);

        acc0 = __builtin_amdgcn_wmma_f32_16x16x32_bf16(false, fa0.v, false, fb.v,
                                                       (short)0, acc0, false, false);
        acc1 = __builtin_amdgcn_wmma_f32_16x16x32_bf16(false, fa1.v, false, fb.v,
                                                       (short)0, acc1, false, false);
        acc2 = __builtin_amdgcn_wmma_f32_16x16x32_bf16(false, fa2.v, false, fb.v,
                                                       (short)0, acc2, false, false);
        acc3 = __builtin_amdgcn_wmma_f32_16x16x32_bf16(false, fa3.v, false, fb.v,
                                                       (short)0, acc3, false, false);

        wg_barrier_reads_done();       // DScnt drained, then signal
    }

    const int outcol = n0 + nsub * 16 + r;
    const float bv = bias[e * OUT_ + outcol];
    const int m0 = mq * 64;
    store_tile(acc0, m0,      hh, mlen, sPairs, out, outcol, bv);
    store_tile(acc1, m0 + 16, hh, mlen, sPairs, out, outcol, bv);
    store_tile(acc2, m0 + 32, hh, mlen, sPairs, out, outcol, bv);
    store_tile(acc3, m0 + 48, hh, mlen, sPairs, out, outcol, bv);
}

extern "C" void kernel_launch(void* const* d_in, const int* in_sizes, int n_in,
                              void* d_out, int out_size, void* d_ws, size_t ws_size,
                              hipStream_t stream) {
    (void)in_sizes; (void)n_in; (void)out_size; (void)ws_size;
    const float* feat = (const float*)d_in[0];   // [B, IN]
    const float* W    = (const float*)d_in[1];   // [E, OUT, IN]
    const float* bias = (const float*)d_in[2];   // [E, OUT]
    const float* act  = (const float*)d_in[3];   // [B, E]
    float* out = (float*)d_out;                  // [B, TOPK, OUT]

    int* counts = (int*)d_ws;                    // 32 ints
    int* pairs  = (int*)((char*)d_ws + 1024);    // E_*B_ ints (~256 KB)

    zero_counts<<<1, 32, 0, stream>>>(counts);
    route_kernel<<<(B_ + 255) / 256, 256, 0, stream>>>(act, counts, pairs);

    dim3 grid(E_, OUT_ / 64, B_ / 256);
    moe_gemm<<<grid, 512, 0, stream>>>(feat, W, bias, counts, pairs, out);
}
